// ModulatedDeformConv2dFastv2Pack_67439576481986
// MI455X (gfx1250) — matrix operand order repaired and checked by reading, not compiled, not run on hardware
//
#include <hip/hip_runtime.h>

// ---------------------------------------------------------------------------
// Modulated deformable conv v2 forward for MI455X (gfx1250, wave32, WMMA).
//   x      : (8, 64, 128, 128) f32
//   w_off  : (27, 64, 3, 3)    f32   (flat = [27][576], K = c*9 + kh*3 + kw)
//   b_off  : (27,)             f32
//   w_conv : (64, 576)         f32   (K = c*9 + tap  -- same K ordering)
//   out    : (8, 64, 128, 128) f32
//
// Kernel 0 pre-converts both weight matrices to bf16 (once, in workspace).
// Kernels 1/2 stage bf16 weight panels into LDS with
// GLOBAL_LOAD_ASYNC_TO_LDS_B128 (ASYNCcnt) and run the two GEMMs on
// v_wmma_f32_16x16x32_bf16 with fp32 accumulation.
// ---------------------------------------------------------------------------

#define BATCH 8
#define CIN   64
#define Hh    128
#define Ww    128
#define HW    (Hh * Ww)          // 16384
#define TAPS  9
#define KDIM  (CIN * TAPS)       // 576
#define KHALF (KDIM / 2)         // 288
#define OFFC  27
#define OC    64

typedef __attribute__((ext_vector_type(16))) __bf16       v16bf;
typedef __attribute__((ext_vector_type(8)))  float        v8f;
typedef __attribute__((ext_vector_type(4)))  float        v4f;
typedef __attribute__((ext_vector_type(4)))  int          v4i;
typedef __attribute__((ext_vector_type(4)))  unsigned int v4u;

union Frag16 {
  v16bf          bf;
  unsigned short us[16];
  v4u            q[2];
};

__device__ __forceinline__ unsigned short f2bf(float f) {
  // round-to-nearest-even f32 -> bf16
  unsigned int u = __float_as_uint(f);
  u += 0x7FFFu + ((u >> 16) & 1u);
  return (unsigned short)(u >> 16);
}

// Async-copy one K-half weight panel (rows x KHALF bf16, row stride KDIM in
// global) into LDS via GLOBAL_LOAD_ASYNC_TO_LDS_B128.  16-B chunks; both
// global and LDS addresses are 16-B aligned (KHALF*2 = 576 = 36*16).
__device__ __forceinline__ void async_stage_panel(
    const unsigned short* __restrict__ gsrc, unsigned short* lds,
    int rows, int kstart, int tid)
{
  const int chunks = rows * (KHALF * 2 / 16);            // 36 chunks per row
  const unsigned int lds_base = (unsigned int)(size_t)lds;  // LDS offset = addr[31:0]
  for (int q = tid; q < chunks; q += 128) {
    int m = q / 36;
    int c = q - m * 36;
    unsigned int ldsoff = lds_base + (unsigned int)(q << 4);
    int goff = (m * KDIM + kstart) * 2 + (c << 4);       // byte offset vs gsrc
    asm volatile("global_load_async_to_lds_b128 %0, %1, %2"
                 :: "v"(ldsoff), "v"(goff), "s"(gsrc)
                 : "memory");
  }
}

__device__ __forceinline__ void wait_async_and_barrier() {
  asm volatile("s_wait_asynccnt 0x0" ::: "memory");
  __syncthreads();
}

// ---------------------------------------------------------------------------
// Kernel 0: one-time fp32 -> bf16 conversion of both weight matrices.
//   wo_bf : [32][576] (rows 27..31 zero)      wc_bf : [64][576]
// ---------------------------------------------------------------------------
__global__ __launch_bounds__(256) void dcn_convert_weights(
    const float* __restrict__ w_off, const float* __restrict__ w_conv,
    unsigned short* __restrict__ wo_bf, unsigned short* __restrict__ wc_bf)
{
  const int i = blockIdx.x * 256 + threadIdx.x;
  if (i < 32 * KDIM)
    wo_bf[i] = (i < OFFC * KDIM) ? f2bf(w_off[i]) : (unsigned short)0;
  if (i < OC * KDIM)
    wc_bf[i] = f2bf(w_conv[i]);
}

// ---------------------------------------------------------------------------
// Kernel 1: 3x3 offset conv via WMMA, then transform to px / py / mask planes.
// Grid: 8 * 256 blocks, 128 threads (4 waves x 16 pixels). M = 27 (padded 32).
// ---------------------------------------------------------------------------
__global__ __launch_bounds__(128) void dcn_offset_wmma(
    const float* __restrict__ x, const unsigned short* __restrict__ wo_bf,
    const float* __restrict__ b_off,
    float* __restrict__ pxb, float* __restrict__ pyb, float* __restrict__ mbuf)
{
  __shared__ unsigned short wlds[32 * KHALF];  // 18432 B (bf16 weight K-half)

  const int tid  = threadIdx.x;
  const int lane = tid & 31;
  const int wv   = tid >> 5;
  const int b    = blockIdx.x >> 8;            // 256 pixel tiles per batch image
  const int tile = blockIdx.x & 255;

  const int p  = tile * 64 + wv * 16 + (lane & 15);  // this lane's pixel (N col)
  const int ph = p >> 7, pw = p & 127;
  const int hi = lane >> 4;                          // lane half (K split)
  const float* xb = x + (size_t)b * (CIN * HW);

  v8f z = {0.f, 0.f, 0.f, 0.f, 0.f, 0.f, 0.f, 0.f};
  v8f acc0 = z, acc1 = z;
  const v4u* wq = (const v4u*)wlds;

  auto compute_half = [&](int kstart) {
    for (int kk = 0; kk < KHALF; kk += 32) {
      const int kb = kstart + kk;
      // --- B fragment: im2col column for this pixel. B layout: K = hi*16 + e
      Frag16 fb;
      const int k0 = kb + hi * 16;
#pragma unroll
      for (int e = 0; e < 16; ++e) {
        int k  = k0 + e;
        int c  = k / 9;
        int t  = k - c * 9;
        int kh = t / 3;
        int kw = t - kh * 3;
        int yy = ph + kh - 1;
        int xx = pw + kw - 1;
        int ok = (yy >= 0) & (yy < Hh) & (xx >= 0) & (xx < Ww);
        int yc = min(max(yy, 0), Hh - 1);
        int xc = min(max(xx, 0), Ww - 1);
        float v = xb[c * HW + yc * Ww + xc];
        fb.us[e] = f2bf(ok ? v : 0.0f);
      }
      // --- A fragments from LDS. A layout: K = hi*16 + e
      Frag16 fa0, fa1;
      const int m0 = lane & 15;
      int o0 = (m0 * KHALF + kk + hi * 16) >> 3;       // v4u units (8 bf16)
      fa0.q[0] = wq[o0];  fa0.q[1] = wq[o0 + 1];
      int o1 = ((m0 + 16) * KHALF + kk + hi * 16) >> 3;
      fa1.q[0] = wq[o1];  fa1.q[1] = wq[o1 + 1];

      acc0 = __builtin_amdgcn_wmma_f32_16x16x32_bf16(false, fa0.bf, false, fb.bf,
                                                     (short)0, acc0, false, false);
      acc1 = __builtin_amdgcn_wmma_f32_16x16x32_bf16(false, fa1.bf, false, fb.bf,
                                                     (short)0, acc1, false, false);
    }
  };

  async_stage_panel(wo_bf, wlds, 32, 0, tid);
  wait_async_and_barrier();
  compute_half(0);
  __syncthreads();                              // all waves done reading half 0
  async_stage_panel(wo_bf, wlds, 32, KHALF, tid);
  wait_async_and_barrier();
  compute_half(KHALF);

  // Epilogue: C/D layout -> oc = f*16 + hi*8 + r, pixel = lane&15 column.
  const float fx = (float)pw, fy = (float)ph;
#pragma unroll
  for (int f = 0; f < 2; ++f) {
    v8f a = f ? acc1 : acc0;
#pragma unroll
    for (int r = 0; r < 8; ++r) {
      int oc = f * 16 + hi * 8 + r;
      if (oc < OFFC) {
        float v = a[r] + b_off[oc];
        int g = oc / 9;
        int t = oc - g * 9;
        size_t o = (size_t)(b * TAPS + t) * HW + p;
        if (g == 0)      pxb[o]  = v + fx + (float)(t % 3 - 1);   // abs x coord
        else if (g == 1) pyb[o]  = v + fy + (float)(t / 3 - 1);   // abs y coord
        else             mbuf[o] = 1.0f / (1.0f + __expf(-v));    // sigmoid
      }
    }
  }
}

// ---------------------------------------------------------------------------
// Kernel 2: bilinear sampling fused with the 576->64 channel GEMM via WMMA.
// Weight staging is async and overlapped with the descriptor build.
// ---------------------------------------------------------------------------
__global__ __launch_bounds__(128) void dcn_sample_gemm_wmma(
    const float* __restrict__ x, const unsigned short* __restrict__ wc_bf,
    const float* __restrict__ pxb, const float* __restrict__ pyb,
    const float* __restrict__ mbuf, float* __restrict__ out)
{
  __shared__ unsigned short wlds[OC * KHALF];  // 36864 B (bf16 w_conv K-half)
  __shared__ v4f dwt[TAPS][64];                //  9216 B bilinear weights
  __shared__ v4i didx[TAPS][64];               //  9216 B corner indices

  const int tid   = threadIdx.x;
  const int lane  = tid & 31;
  const int wv    = tid >> 5;
  const int b     = blockIdx.x >> 8;
  const int tile  = blockIdx.x & 255;
  const int pbase = tile * 64;

  // Kick off async staging of weight K-half 0 first...
  async_stage_panel(wc_bf, wlds, OC, 0, tid);

  // ...and overlap it with the bilinear-descriptor build (64 pixels x 9 taps).
  for (int i = tid; i < TAPS * 64; i += 128) {
    int t  = i >> 6, pl = i & 63;
    size_t o = (size_t)(b * TAPS + t) * HW + pbase + pl;
    float px = pxb[o], py = pyb[o], m = mbuf[o];
    float x0f = floorf(px), y0f = floorf(py);
    float wx1 = px - x0f, wy1 = py - y0f;
    float wx0 = 1.0f - wx1, wy0 = 1.0f - wy1;
    int x0 = (int)x0f, y0 = (int)y0f;
    int x1 = x0 + 1,  y1 = y0 + 1;
    float vx0 = (x0 >= 0 && x0 < Ww) ? 1.0f : 0.0f;
    float vx1 = (x1 >= 0 && x1 < Ww) ? 1.0f : 0.0f;
    float vy0 = (y0 >= 0 && y0 < Hh) ? 1.0f : 0.0f;
    float vy1 = (y1 >= 0 && y1 < Hh) ? 1.0f : 0.0f;
    int x0c = min(max(x0, 0), Ww - 1), x1c = min(max(x1, 0), Ww - 1);
    int y0c = min(max(y0, 0), Hh - 1), y1c = min(max(y1, 0), Hh - 1);
    v4f w;
    w.x = wy0 * wx0 * m * vy0 * vx0;
    w.y = wy0 * wx1 * m * vy0 * vx1;
    w.z = wy1 * wx0 * m * vy1 * vx0;
    w.w = wy1 * wx1 * m * vy1 * vx1;
    v4i ii;
    ii.x = y0c * Ww + x0c;
    ii.y = y0c * Ww + x1c;
    ii.z = y1c * Ww + x0c;
    ii.w = y1c * Ww + x1c;
    dwt[t][pl]  = w;
    didx[t][pl] = ii;
  }

  const int pl = wv * 16 + (lane & 15);
  const int hi = lane >> 4;
  const float* xb = x + (size_t)b * (CIN * HW);
  const v4u* wq = (const v4u*)wlds;

  v8f z = {0.f, 0.f, 0.f, 0.f, 0.f, 0.f, 0.f, 0.f};
  v8f acc[4] = {z, z, z, z};

  auto compute_half = [&](int kstart) {
    for (int kk = 0; kk < KHALF; kk += 32) {
      const int kb = kstart + kk;
      // --- B fragment: sampled y[k, pixel] values (K = hi*16 + e)
      Frag16 fbv;
      const int k0 = kb + hi * 16;
#pragma unroll
      for (int e = 0; e < 16; ++e) {
        int k = k0 + e;
        int c = k / 9;
        int t = k - c * 9;
        v4f w  = dwt[t][pl];
        v4i ii = didx[t][pl];
        const float* xp = xb + c * HW;
        float v = w.x * xp[ii.x] + w.y * xp[ii.y] +
                  w.z * xp[ii.z] + w.w * xp[ii.w];
        fbv.us[e] = f2bf(v);
      }
      // --- 4 M-tiles cover all 64 output channels, reusing one B fragment
#pragma unroll
      for (int mt = 0; mt < 4; ++mt) {
        Frag16 fa;
        int m  = mt * 16 + (lane & 15);
        int o0 = (m * KHALF + kk + hi * 16) >> 3;
        fa.q[0] = wq[o0];  fa.q[1] = wq[o0 + 1];
        acc[mt] = __builtin_amdgcn_wmma_f32_16x16x32_bf16(false, fa.bf, false, fbv.bf,
                                                          (short)0, acc[mt], false, false);
      }
    }
  };

  wait_async_and_barrier();
  compute_half(0);
  __syncthreads();                              // all waves done reading half 0
  async_stage_panel(wc_bf, wlds, OC, KHALF, tid);
  wait_async_and_barrier();
  compute_half(KHALF);

  // Store: out[b][oc][p], oc = mt*16 + hi*8 + r, p = pbase + pl.
  float* ob = out + (size_t)b * (OC * HW) + pbase + pl;
#pragma unroll
  for (int mt = 0; mt < 4; ++mt) {
#pragma unroll
    for (int r = 0; r < 8; ++r) {
      ob[(size_t)(mt * 16 + hi * 8 + r) * HW] = acc[mt][r];
    }
  }
}

// ---------------------------------------------------------------------------
extern "C" void kernel_launch(void* const* d_in, const int* in_sizes, int n_in,
                              void* d_out, int out_size, void* d_ws, size_t ws_size,
                              hipStream_t stream) {
  (void)in_sizes; (void)n_in; (void)out_size; (void)ws_size;
  const float* x      = (const float*)d_in[0];
  const float* w_off  = (const float*)d_in[1];
  const float* b_off  = (const float*)d_in[2];
  const float* w_conv = (const float*)d_in[3];
  float* out = (float*)d_out;

  // workspace layout:
  //   px, py, mask planes : 3 x (8, 9, 128, 128) f32   (14.2 MB)
  //   wo_bf [32][576] bf16, wc_bf [64][576] bf16       (0.11 MB)
  float* pxb = (float*)d_ws;
  float* pyb = pxb + (size_t)BATCH * TAPS * HW;
  float* mb  = pyb + (size_t)BATCH * TAPS * HW;
  unsigned short* wo_bf = (unsigned short*)(mb + (size_t)BATCH * TAPS * HW);
  unsigned short* wc_bf = wo_bf + 32 * KDIM;

  dcn_convert_weights<<<dim3((OC * KDIM + 255) / 256), dim3(256), 0, stream>>>(
      w_off, w_conv, wo_bf, wc_bf);

  const int blocks = BATCH * (HW / 64);  // 2048
  dcn_offset_wmma<<<dim3(blocks), dim3(128), 0, stream>>>(x, wo_bf, b_off, pxb, pyb, mb);
  dcn_sample_gemm_wmma<<<dim3(blocks), dim3(128), 0, stream>>>(x, wc_bf, pxb, pyb, mb, out);
}
